// BosonSamplerTorch_59906203844741
// MI455X (gfx1250) — matrix operand order repaired and checked by reading, not compiled, not run on hardware
//
#include <hip/hip_runtime.h>

typedef __attribute__((ext_vector_type(16))) _Float16 v16h;
typedef __attribute__((ext_vector_type(8)))  float    v8f;

#define N_PH 20
#define GROUPS_TOTAL   (1u << (N_PH - 4))      // 65536 groups of 16 sign vectors
#define BLOCKS         1024
#define THREADS        256
#define WAVES_PER_BLOCK (THREADS / 32)
#define TOTAL_WAVES    (BLOCKS * WAVES_PER_BLOCK)
#define GROUPS_PER_WAVE (GROUPS_TOTAL / TOTAL_WAVES)   // 8

static __device__ __forceinline__ v8f wmma_acc(v16h a, v16h b, v8f c) {
  // D = A x B + C, 16x16x32 f16 inputs, f32 accumulate
  return __builtin_amdgcn_wmma_f32_16x16x32_f16(
      /*neg_a=*/false, a, /*neg_b=*/false, b,
      /*c_mod=*/(short)0, c, /*reuse_a=*/false, /*reuse_b=*/false);
}

__global__ void __launch_bounds__(THREADS)
glynn_wmma_kernel(const float* __restrict__ Ar,
                  const float* __restrict__ Ai,
                  float* __restrict__ ws) {
  const int lane = threadIdx.x & 31;
  const int lrow = lane & 15;          // M (A-operand) / N sign-row (B,C)
  const int grp  = (lane >> 4) & 1;    // lane half selects K subrange

  __builtin_prefetch(Ar, 0, 0);        // global_prefetch_b8: pin A toward caches
  __builtin_prefetch(Ai, 0, 0);

  // ---- Build wave-invariant WMMA A-operands: A^T in the ISA 16-bit A layout.
  // Element e = 2v+h of v16h <-> VGPR v, half h.  K map (ISA 7.12.2):
  //   lanes 0-15 : v<4 -> K=2v+h       ; v>=4 -> K=8+2v+h   (K in 0..7, 16..23)
  //   lanes 16-31: same +8             (K in 8..15, 24..31)
  // hi/lo split gives ~f32 accuracy with f16 WMMA (signs are exact in f16).
  v16h aHi[2][2], aLo[2][2];           // [tile(cols 0-15 / 16-31)][re/im]
#pragma unroll
  for (int t = 0; t < 2; ++t) {
#pragma unroll
    for (int e = 0; e < 16; ++e) {
      const int v = e >> 1, h = e & 1;
      const int k = ((v < 4) ? (2 * v + h) : (8 + 2 * v + h)) + grp * 8;
      const int m = lrow + 16 * t;     // global column of A
      float vr = 0.0f, vi = 0.0f;
      if (k < N_PH && m < N_PH) {      // pad K>=20 and cols>=20 with zeros
        vr = Ar[k * N_PH + m];
        vi = Ai[k * N_PH + m];
      }
      const _Float16 rh = (_Float16)vr;
      const _Float16 ih = (_Float16)vi;
      aHi[t][0][e] = rh; aLo[t][0][e] = (_Float16)(vr - (float)rh);
      aHi[t][1][e] = ih; aLo[t][1][e] = (_Float16)(vi - (float)ih);
    }
  }

  const int waveId = blockIdx.x * WAVES_PER_BLOCK + (threadIdx.x >> 5);
  float accR = 0.0f, accI = 0.0f;

  for (int g = 0; g < GROUPS_PER_WAVE; ++g) {
    const unsigned base = (((unsigned)waveId * GROUPS_PER_WAVE) + g) << 4;
    const unsigned idx  = base + (unsigned)lrow;   // this lane's sign-vector id

    // ---- B-operand: signs^T (32x16 f16). B layout: lane n holds column n;
    // same K map as A: lanes 0-15 K=0..7,16..23; lanes 16-31 K=8..15,24..31.
    v16h b;
#pragma unroll
    for (int h = 0; h < 16; ++h) {
      const int k = ((h < 8) ? h : (8 + h)) + grp * 8;
      const float s = (k < N_PH) ? (((idx >> k) & 1u) ? 1.0f : -1.0f) : 0.0f;
      b[h] = (_Float16)s;
    }

    // ---- 8 WMMAs: S^T tiles, real & imag, hi+lo chained into one f32 acc.
    v8f cR0 = {}; cR0 = wmma_acc(aHi[0][0], b, cR0); cR0 = wmma_acc(aLo[0][0], b, cR0);
    v8f cI0 = {}; cI0 = wmma_acc(aHi[0][1], b, cI0); cI0 = wmma_acc(aLo[0][1], b, cI0);
    v8f cR1 = {}; cR1 = wmma_acc(aHi[1][0], b, cR1); cR1 = wmma_acc(aLo[1][0], b, cR1);
    v8f cI1 = {}; cI1 = wmma_acc(aHi[1][1], b, cI1); cI1 = wmma_acc(aLo[1][1], b, cI1);

    // ---- Per-lane complex product. C layout: lanes 0-15 hold cols v (tile0)
    // and 16+v (tile1, v<4 valid); lanes 16-31 hold cols 8+v (tile1 slots are
    // zero-padded there -> multiply identity instead).
    float pr = cR0[0], pi = cI0[0];
#pragma unroll
    for (int v = 1; v < 8; ++v) {
      const float qr = cR0[v], qi = cI0[v];
      const float nr = pr * qr - pi * qi;
      const float ni = pr * qi + pi * qr;
      pr = nr; pi = ni;
    }
    const bool lowHalf = (lane < 16);
#pragma unroll
    for (int v = 0; v < 4; ++v) {
      const float qr = lowHalf ? cR1[v] : 1.0f;
      const float qi = lowHalf ? cI1[v] : 0.0f;
      const float nr = pr * qr - pi * qi;
      const float ni = pr * qi + pi * qr;
      pr = nr; pi = ni;
    }
    // combine the two half-row products across lane halves
    const float orr = __shfl_xor(pr, 16, 32);
    const float oii = __shfl_xor(pi, 16, 32);
    const float fr = pr * orr - pi * oii;
    const float fi = pr * oii + pi * orr;

    // Glynn parity coefficient; only the low half accumulates (mirror = dup)
    const float w =
        lowHalf ? ((__popc(idx) & 1) ? -1.0f : 1.0f) : 0.0f;
    accR = fmaf(w, fr, accR);
    accI = fmaf(w, fi, accI);
  }

  // ---- Deterministic reduction: wave shfl-xor, then LDS across 8 waves.
#pragma unroll
  for (int off = 16; off >= 1; off >>= 1) {
    accR += __shfl_xor(accR, off, 32);
    accI += __shfl_xor(accI, off, 32);
  }
  __shared__ float sR[WAVES_PER_BLOCK], sI[WAVES_PER_BLOCK];
  const int w = threadIdx.x >> 5;
  if (lane == 0) { sR[w] = accR; sI[w] = accI; }
  __syncthreads();
  if (threadIdx.x == 0) {
    float r = 0.0f, i = 0.0f;
#pragma unroll
    for (int j = 0; j < WAVES_PER_BLOCK; ++j) { r += sR[j]; i += sI[j]; }
    ws[2 * blockIdx.x + 0] = r;   // every slot rewritten every launch
    ws[2 * blockIdx.x + 1] = i;
  }
}

__global__ void __launch_bounds__(256)
glynn_finalize(const float* __restrict__ ws, float* __restrict__ out,
               int nPartials) {
  __shared__ float sR[256], sI[256];
  float r = 0.0f, i = 0.0f;
  for (int t = threadIdx.x; t < nPartials; t += 256) {
    r += ws[2 * t + 0];
    i += ws[2 * t + 1];
  }
  sR[threadIdx.x] = r; sI[threadIdx.x] = i;
  __syncthreads();
  for (int off = 128; off > 0; off >>= 1) {
    if ((int)threadIdx.x < off) {
      sR[threadIdx.x] += sR[threadIdx.x + off];
      sI[threadIdx.x] += sI[threadIdx.x + off];
    }
    __syncthreads();
  }
  if (threadIdx.x == 0) {
    const float pr = sR[0] * 0x1p-19f;   // 2^(1-n), n=20
    const float pi = sI[0] * 0x1p-19f;
    out[0] = pr * pr + pi * pi;          // |Perm(A)|^2
  }
}

extern "C" void kernel_launch(void* const* d_in, const int* in_sizes, int n_in,
                              void* d_out, int out_size, void* d_ws, size_t ws_size,
                              hipStream_t stream) {
  const float* Ar = (const float*)d_in[0];
  const float* Ai = (const float*)d_in[1];
  float* out = (float*)d_out;
  float* ws  = (float*)d_ws;   // needs 2*BLOCKS floats = 8 KB

  glynn_wmma_kernel<<<BLOCKS, THREADS, 0, stream>>>(Ar, Ai, ws);
  glynn_finalize<<<1, 256, 0, stream>>>(ws, out, BLOCKS);
}
